// ConditionalDecoder_90048284328474
// MI455X (gfx1250) — matrix operand order repaired and checked by reading, not compiled, not run on hardware
//
#include <hip/hip_runtime.h>

#define BB 2048
#define HID 256
#define COND 128
#define NCH 64
#define MAXLEN 128
#define RR 256

typedef __attribute__((ext_vector_type(16))) __bf16 v16bf;
typedef __attribute__((ext_vector_type(8)))  float  v8f;
typedef __attribute__((ext_vector_type(4)))  unsigned int u32x4;

union Frag {
    v16bf v;
    u32x4 q[2];
};

__device__ __forceinline__ unsigned short f32_bf16(float f) {
    unsigned int u = __float_as_uint(f);
    u += 0x7FFFu + ((u >> 16) & 1u);   // round-to-nearest-even
    return (unsigned short)(u >> 16);
}

__device__ __forceinline__ float sigmoid_f(float x) {
    return 1.0f / (1.0f + __expf(-x));
}

// One 16x16 output tile over K=256: 8 x v_wmma_f32_16x16x32_bf16.
// wrow points at this lane's weight row (lane ln -> output column), k-chunk base.
__device__ __forceinline__ v8f gemm_tile(const Frag (&a)[8],
                                         const unsigned short* __restrict__ wrow) {
    v8f acc = {};
    #pragma unroll
    for (int k = 0; k < 8; ++k) {
        Frag b;   // B layout: lanes 0-15 col n K k0..k0+15, lanes 16-31 K k0+16..k0+31
        const u32x4* p = (const u32x4*)(wrow + k * 32);
        b.q[0] = p[0];
        b.q[1] = p[1];
        acc = __builtin_amdgcn_wmma_f32_16x16x32_bf16(
                false, a[k].v, false, b.v, (short)0, acc, false, false);
    }
    return acc;
}

// ---- kernel 1: build bf16 weight blobs ----------------------------------------
// Wfold (1024x256): rows 0..255  = W_ih_r + W_hh_r
//                   rows 256..511 = W_ih_z + W_hh_z
//                   rows 512..767 = W_ih_n
//                   rows 768..1023= W_hh_n
// Whh0  (768x256):  bf16(W_hh)           (step-0 path: input is zeros)
// WoutB (64x256):   bf16(W_out)
__global__ void cvt_weights_kernel(const float* __restrict__ W_ih,
                                   const float* __restrict__ W_hh,
                                   const float* __restrict__ W_out,
                                   unsigned short* __restrict__ Wfold,
                                   unsigned short* __restrict__ Whh0,
                                   unsigned short* __restrict__ WoutB) {
    int i = blockIdx.x * blockDim.x + threadIdx.x;
    const int NF = 1024 * 256, NH = 768 * 256, NO = NCH * 256;
    if (i < NF) {
        int r = i >> 8;
        float v;
        if (r < 512)       v = W_ih[i] + W_hh[i];
        else if (r < 768)  v = W_ih[i];
        else               v = W_hh[i - 256 * 256];
        Wfold[i] = f32_bf16(v);
    } else if (i < NF + NH) {
        Whh0[i - NF] = f32_bf16(W_hh[i - NF]);
    } else if (i < NF + NH + NO) {
        WoutB[i - NF - NH] = f32_bf16(W_out[i - NF - NH]);
    }
}

// ---- kernel 2: h0 = [z | x_cond] @ W_lh^T + b_lh ------------------------------
__global__ void h0_kernel(const float* __restrict__ z,
                          const float* __restrict__ xc,
                          const float* __restrict__ W_lh,
                          const float* __restrict__ b_lh,
                          float* __restrict__ h0) {
    int i = blockIdx.x * blockDim.x + threadIdx.x;   // b*256 + o
    int b = i >> 8, o = i & 255;
    const float* w  = W_lh + (size_t)o * (HID + COND);
    const float* zb = z  + (size_t)b * HID;
    const float* xb = xc + (size_t)b * COND;
    float acc = b_lh[o];
    #pragma unroll 4
    for (int k = 0; k < HID; ++k)  acc += zb[k] * w[k];
    #pragma unroll 4
    for (int k = 0; k < COND; ++k) acc += xb[k] * w[HID + k];
    h0[i] = acc;
}

// ---- kernel 3: persistent GRU decoder, 16 batch rows / block ------------------
__launch_bounds__(256)
__global__ void decoder_kernel(const float* __restrict__ h0,
                               const unsigned short* __restrict__ Wfold,
                               const unsigned short* __restrict__ Whh0,
                               const unsigned short* __restrict__ WoutB,
                               const float* __restrict__ b_ih,
                               const float* __restrict__ b_hh,
                               const float* __restrict__ b_out,
                               float* __restrict__ out) {
    __shared__ unsigned short Ah[16 * 256];   // 8 KB: bf16 hidden state tile

    const int tid  = threadIdx.x;
    const int lane = tid & 31;
    const int wv   = tid >> 5;       // wave 0..7 (wave32)
    const int hh   = lane >> 4;      // half-wave 0/1
    const int ln   = lane & 15;
    const int wgBase = blockIdx.x * 16;

    for (int i = tid; i < 16 * 256; i += 256)
        Ah[i] = f32_bf16(h0[(size_t)(wgBase + (i >> 8)) * 256 + (i & 255)]);

    // Per-wave persistent state: this wave owns hidden columns of j-tiles 2w, 2w+1.
    float hprev[2][8];
    float biR[2], biZ[2], biN[2], bhR[2], bhZ[2], bhN[2];
    #pragma unroll
    for (int jt = 0; jt < 2; ++jt) {
        const int col = (wv * 2 + jt) * 16 + ln;
        #pragma unroll
        for (int i = 0; i < 8; ++i)
            hprev[jt][i] = h0[(size_t)(wgBase + hh * 8 + i) * 256 + col];
        biR[jt] = b_ih[col];       bhR[jt] = b_hh[col];
        biZ[jt] = b_ih[256 + col]; bhZ[jt] = b_hh[256 + col];
        biN[jt] = b_ih[512 + col]; bhN[jt] = b_hh[512 + col];
    }
    const float outBias = (wv < 4) ? b_out[wv * 16 + ln] : 0.0f;
    __syncthreads();

    for (int t = 0; t < MAXLEN; ++t) {
        // A fragments (h_{t-1}, bf16) from LDS, contiguous k layout, same as B:
        // lanes 0-15: K k0+0..15 ; lanes 16-31: K k0+16..31
        Frag a[8];
        #pragma unroll
        for (int k = 0; k < 8; ++k) {
            const u32x4* p = (const u32x4*)(Ah + ln * 256 + k * 32 + hh * 16);
            a[k].q[0] = p[0];
            a[k].q[1] = p[1];
        }
        __syncthreads();   // all waves captured A; Ah is now writable

        #pragma unroll
        for (int jt = 0; jt < 2; ++jt) {
            const int col = (wv * 2 + jt) * 16 + ln;   // this lane's output column
            if (t == 0) {
                // x = 0  =>  gi = b_ih ; gh = h0 @ W_hh^T + b_hh
                const unsigned short* w0 = Whh0 + (size_t)col * 256 + hh * 16;
                v8f aR = gemm_tile(a, w0);
                __builtin_amdgcn_sched_barrier(0);
                v8f aZ = gemm_tile(a, w0 + 256 * 256);
                __builtin_amdgcn_sched_barrier(0);
                v8f aN = gemm_tile(a, w0 + 512 * 256);
                __builtin_amdgcn_sched_barrier(0);
                #pragma unroll
                for (int i = 0; i < 8; ++i) {
                    float r = sigmoid_f(biR[jt] + aR[i] + bhR[jt]);
                    float u = sigmoid_f(biZ[jt] + aZ[i] + bhZ[jt]);
                    float n = tanhf(biN[jt] + r * (aN[i] + bhN[jt]));
                    float hnew = (1.0f - u) * n + u * hprev[jt][i];
                    hprev[jt][i] = hnew;
                    Ah[(hh * 8 + i) * 256 + col] = f32_bf16(hnew);
                }
            } else {
                // x == h  =>  r/z fold onto summed weights; n stays split
                const unsigned short* wf = Wfold + (size_t)col * 256 + hh * 16;
                v8f aR = gemm_tile(a, wf);                   // (W_ih_r+W_hh_r)
                __builtin_amdgcn_sched_barrier(0);
                v8f aZ = gemm_tile(a, wf + 256 * 256);       // (W_ih_z+W_hh_z)
                __builtin_amdgcn_sched_barrier(0);
                v8f aI = gemm_tile(a, wf + 512 * 256);       // W_ih_n
                __builtin_amdgcn_sched_barrier(0);
                v8f aH = gemm_tile(a, wf + 768 * 256);       // W_hh_n
                __builtin_amdgcn_sched_barrier(0);
                #pragma unroll
                for (int i = 0; i < 8; ++i) {
                    float r = sigmoid_f(aR[i] + biR[jt] + bhR[jt]);
                    float u = sigmoid_f(aZ[i] + biZ[jt] + bhZ[jt]);
                    float n = tanhf(aI[i] + biN[jt] + r * (aH[i] + bhN[jt]));
                    float hnew = (1.0f - u) * n + u * hprev[jt][i];
                    hprev[jt][i] = hnew;
                    Ah[(hh * 8 + i) * 256 + col] = f32_bf16(hnew);
                }
            }
        }
        __syncthreads();   // Ah now holds h_t

        // logits_t = h_t @ W_out^T + b_out : 4 N-tiles, waves 0..3
        if (wv < 4) {
            Frag ao[8];
            #pragma unroll
            for (int k = 0; k < 8; ++k) {
                const u32x4* p = (const u32x4*)(Ah + ln * 256 + k * 32 + hh * 16);
                ao[k].q[0] = p[0];
                ao[k].q[1] = p[1];
            }
            v8f acc = gemm_tile(ao, WoutB + (size_t)(wv * 16 + ln) * 256 + hh * 16);
            const int col = wv * 16 + ln;
            #pragma unroll
            for (int i = 0; i < 8; ++i) {
                const int rowM = hh * 8 + i;
                out[((size_t)(wgBase + rowM) * MAXLEN + t) * NCH + col] = acc[i] + outBias;
            }
        }
        // next iteration's A-loads (reads of Ah) are safe alongside the out-GEMM;
        // writes to Ah only happen after the post-load barrier of step t+1.
    }
}

extern "C" void kernel_launch(void* const* d_in, const int* in_sizes, int n_in,
                              void* d_out, int out_size, void* d_ws, size_t ws_size,
                              hipStream_t stream) {
    const float* z     = (const float*)d_in[0];
    const float* xcond = (const float*)d_in[1];
    const float* W_lh  = (const float*)d_in[2];
    const float* b_lh  = (const float*)d_in[3];
    const float* W_ih  = (const float*)d_in[4];
    const float* W_hh  = (const float*)d_in[5];
    const float* b_ih  = (const float*)d_in[6];
    const float* b_hh  = (const float*)d_in[7];
    const float* W_out = (const float*)d_in[8];
    const float* b_out = (const float*)d_in[9];
    float* out = (float*)d_out;

    char* ws = (char*)d_ws;
    float*          h0    = (float*)ws;                               // 2 MB
    unsigned short* Wfold = (unsigned short*)(ws + (size_t)BB * RR * 4);
    unsigned short* Whh0  = Wfold + 1024 * 256;                       // +512 KB
    unsigned short* WoutB = Whh0  + 768 * 256;                        // +384 KB (+32 KB)

    const int cvtN = (1024 + 768 + NCH) * 256;
    cvt_weights_kernel<<<(cvtN + 255) / 256, 256, 0, stream>>>(
        W_ih, W_hh, W_out, Wfold, Whh0, WoutB);
    h0_kernel<<<(BB * RR) / 256, 256, 0, stream>>>(z, xcond, W_lh, b_lh, h0);
    decoder_kernel<<<BB / 16, 256, 0, stream>>>(
        h0, Wfold, Whh0, WoutB, b_ih, b_hh, b_out, out);
}
